// MSPyramidAttentionContextModule_16896401343294
// MI455X (gfx1250) — compile-verified, matrix operand order checked by reading, not run on hardware
//
#include <hip/hip_runtime.h>
#include <hip/hip_fp16.h>

// MS Pyramid Attention Context Module for MI455X (gfx1250, wave32, WMMA).
// Pipeline:
//   K0: feats fp32 -> ctx f16 channel-last (B,N,1024), channels [0,512)
//   K1: theta/phi/g 1x1-conv GEMMs (f16 WMMA, fp32 acc); theta scaled by c1^-0.5
//       theta,phi stored transposed (N,256); g stored natural (256,N)
//   K2: flash-style attention per (g,b,64-row block): S=Q K^T streamed in
//       32-col chunks, online softmax, Y^T accumulated in VGPRs via WMMA,
//       global->reg->LDS software pipeline to overlap HBM with WMMA
//   K3: W-conv GEMM -> wy fp32 (channel-major for BN)
//   K4: BN stats (mean, rsqrt(var+eps)) one block/channel (deterministic)
//   K5: BN apply + scatter into ctx channels [512,1024)
//   K6: bottleneck GEMM (512x1024) over ctx -> bco fp32 (channel-major)
//   K7: BN stats, K8: BN apply -> d_out (B,512,N) fp32
// Workspace: 5 * 37,748,736 B + stats (~189 MB), with dead-buffer reuse.

#define DEVINL static __device__ __forceinline__

typedef __attribute__((ext_vector_type(16))) _Float16 v16h;
typedef __attribute__((ext_vector_type(8)))  _Float16 v8h;
typedef __attribute__((ext_vector_type(4)))  _Float16 v4h;
typedef __attribute__((ext_vector_type(8)))  float    v8f;
typedef __attribute__((ext_vector_type(4)))  float    v4f;

#define Bq   8
#define Cq   512
#define Nq   2304
#define Gq   4
#define C1q  256
#define CGq  128
#define OUTq 512

// Load one 16x32 (rows x K) f16 fragment. Rows (A) or cols (B, staged
// [col][k]) are contiguous with `pitch` halfs. Per-lane K pattern follows the
// CDNA5 16-bit A-matrix table: lane<16 -> K {0..7, 16..23}; lane>=16 -> +8.
// Two 16-byte loads per lane (pitch must be a multiple of 8 halfs).
DEVINL v16h load_frag(const _Float16* p, int lane, int pitch) {
  const int row  = lane & 15;
  const int koff = (lane >> 4) << 3;
  const _Float16* r = p + row * pitch + koff;
  v8h lo = *(const v8h*)(r);
  v8h hi = *(const v8h*)(r + 16);
  v16h o;
#pragma unroll
  for (int i = 0; i < 8; ++i) { o[i] = lo[i]; o[i + 8] = hi[i]; }
  return o;
}

DEVINL v8f wmma_f16(v16h a, v16h b, v8f c) {
  return __builtin_amdgcn_wmma_f32_16x16x32_f16(false, a, false, b,
                                                (short)0, c, false, false);
}

DEVINL v4h cvt4(v4f x) {
  v4h h;
#pragma unroll
  for (int i = 0; i < 4; ++i) h[i] = (_Float16)x[i];
  return h;
}

// ---------------------------------------------------------------- K0: pack
__global__ void k0_pack_feats(const float* __restrict__ feats,
                              _Float16* __restrict__ ctx) {
  size_t idx = (size_t)blockIdx.x * blockDim.x + threadIdx.x;
  if (idx >= (size_t)Bq * Cq * Nq) return;
  int n = (int)(idx % Nq);
  int c = (int)((idx / Nq) % Cq);
  int b = (int)(idx / ((size_t)Nq * Cq));
  ctx[((size_t)b * Nq + n) * 1024 + c] = (_Float16)feats[idx];
}

// ------------------------------------------------------- K1: projections
// grid: x = mtile(4) * ntile(36), y = g*B+b (32), z = proj (3); block 128.
__global__ __launch_bounds__(128)
void k1_proj(const float* __restrict__ feats,
             const float* __restrict__ tw, const float* __restrict__ tb,
             const float* __restrict__ pw, const float* __restrict__ pb,
             const float* __restrict__ gw, const float* __restrict__ gbias,
             _Float16* __restrict__ th_t, _Float16* __restrict__ ph_t,
             _Float16* __restrict__ gx) {
  __shared__ _Float16 lds_a[64 * 40];
  __shared__ _Float16 lds_b[64 * 40];
  const int proj = blockIdx.z;
  const int gb   = blockIdx.y;                 // g*B + b
  const int g = gb >> 3, b = gb & 7;
  const int m0 = (blockIdx.x / 36) * 64;
  const int n0 = (blockIdx.x % 36) * 64;
  const int t = threadIdx.x, wid = t >> 5, lane = t & 31;

  const float* W; const float* bias; float scale;
  if (proj == 0)      { W = tw + (size_t)g * C1q * Cq; bias = tb + g * C1q;    scale = 0.0625f; }
  else if (proj == 1) { W = pw + (size_t)g * C1q * Cq; bias = pb + g * C1q;    scale = 1.0f; }
  else                { W = gw + (size_t)g * C1q * Cq; bias = gbias + g * C1q; scale = 1.0f; }

  v8f acc[4] = {};                             // wave: 16(m) x 64(n)
  const int mw = m0 + wid * 16;

  for (int kk = 0; kk < Cq; kk += 32) {
    // gather phase: 8 b128 global loads in flight per thread
    v4f av[4], bv[4];
#pragma unroll
    for (int e = 0; e < 4; ++e) {
      int flat = (e * 128 + t) * 4;            // A: [i 0..63][j 0..31]
      av[e] = *(const v4f*)(W + (size_t)(m0 + (flat >> 5)) * Cq + kk + (flat & 31));
    }
#pragma unroll
    for (int e = 0; e < 4; ++e) {
      int flat = (e * 128 + t) * 4;            // B: 4 consecutive n, same k
      bv[e] = *(const v4f*)(feats + ((size_t)b * Cq + kk + (flat >> 6)) * Nq
                            + n0 + (flat & 63));
    }
    __syncthreads();
    // scatter phase: cvt + LDS stores
#pragma unroll
    for (int e = 0; e < 4; ++e) {
      int flat = (e * 128 + t) * 4;
      *(v4h*)(lds_a + (flat >> 5) * 40 + (flat & 31)) = cvt4(av[e]);
      int nn = flat & 63, kq = flat >> 6;
      v4h h = cvt4(bv[e]);
      lds_b[(nn + 0) * 40 + kq] = h[0];
      lds_b[(nn + 1) * 40 + kq] = h[1];
      lds_b[(nn + 2) * 40 + kq] = h[2];
      lds_b[(nn + 3) * 40 + kq] = h[3];
    }
    __syncthreads();
    v16h a = load_frag(lds_a + wid * 16 * 40, lane, 40);
#pragma unroll
    for (int j = 0; j < 4; ++j) {
      v16h bb = load_frag(lds_b + j * 16 * 40, lane, 40);
      acc[j] = wmma_f16(a, bb, acc[j]);
    }
  }

  const int nl = lane & 15, mh = (lane >> 4) * 8;
  float bias_v[8];
#pragma unroll
  for (int v = 0; v < 8; ++v) bias_v[v] = bias[mw + mh + v];

  if (proj == 2) {                             // g: natural (256, N)
#pragma unroll
    for (int j = 0; j < 4; ++j) {
      int n = n0 + j * 16 + nl;
#pragma unroll
      for (int v = 0; v < 8; ++v) {
        int m = mw + mh + v;
        gx[((size_t)gb * C1q + m) * Nq + n] = (_Float16)(acc[j][v] + bias_v[v]);
      }
    }
  } else {                                     // theta/phi: transposed (N, 256)
    _Float16* dst = (proj == 0) ? th_t : ph_t;
#pragma unroll
    for (int j = 0; j < 4; ++j) {
      int n = n0 + j * 16 + nl;
#pragma unroll
      for (int v = 0; v < 8; ++v) {
        int m = mw + mh + v;
        dst[((size_t)gb * Nq + n) * C1q + m] =
            (_Float16)((acc[j][v] + bias_v[v]) * scale);
      }
    }
  }
}

// ---------------------------------------------- K2 staging helpers
DEVINL void ld_ph(const _Float16* __restrict__ phb, int nc, int t, v8h (&r)[8]) {
#pragma unroll
  for (int e = 0; e < 8; ++e) {
    int chunk = e * 128 + t;                   // [row 0..31][col8 0..31]
    r[e] = *(const v8h*)(phb + (size_t)(nc + (chunk >> 5)) * C1q + (chunk & 31) * 8);
  }
}
DEVINL void ld_gx(const _Float16* __restrict__ gxb, int nc, int t, v8h (&r)[8]) {
#pragma unroll
  for (int e = 0; e < 8; ++e) {
    int chunk = e * 128 + t;                   // [c 0..255][q 0..3]
    r[e] = *(const v8h*)(gxb + (size_t)(chunk >> 2) * Nq + nc + (chunk & 3) * 8);
  }
}
DEVINL void st_ph(_Float16* lds_ph, int t, const v8h (&r)[8]) {
#pragma unroll
  for (int e = 0; e < 8; ++e) {
    int chunk = e * 128 + t;
    *(v8h*)(lds_ph + (chunk >> 5) * 264 + (chunk & 31) * 8) = r[e];
  }
}
DEVINL void st_gx(_Float16* lds_gx, int t, const v8h (&r)[8]) {
#pragma unroll
  for (int e = 0; e < 8; ++e) {
    int chunk = e * 128 + t;
    *(v8h*)(lds_gx + (chunk >> 2) * 40 + (chunk & 3) * 8) = r[e];
  }
}

// ------------------------------------------- K2: flash attention per stage
// grid: x = m-block (36), y = g*B+b (32); block 128 (4 waves, 16 rows each).
__global__ __launch_bounds__(128)
void k2_attn(const _Float16* __restrict__ th_t,
             const _Float16* __restrict__ ph_t,
             const _Float16* __restrict__ gx,
             _Float16* __restrict__ y_t) {
  __shared__ _Float16 lds_ph[32 * 264];        // [n][c], pitch 264
  __shared__ _Float16 lds_gx[256 * 40];        // [c][n], pitch 40
  __shared__ _Float16 lds_p[4 * 16 * 40];      // per-wave P [m][n]
  const int gb = blockIdx.y;
  const int m0 = blockIdx.x * 64;
  const int t = threadIdx.x, w = t >> 5, lane = t & 31;
  const int mw = m0 + w * 16;

  // Q fragments resident in VGPRs for the whole n-loop (K=256 -> 8 frags).
  const _Float16* thb = th_t + ((size_t)gb * Nq + mw) * C1q;
  v16h ath[8];
#pragma unroll
  for (int s = 0; s < 8; ++s) ath[s] = load_frag(thb + s * 32, lane, C1q);

  v8f yacc[16] = {};                           // 16(m) x 256(c) fp32 acc
  float rmax[8], rsum[8];
#pragma unroll
  for (int v = 0; v < 8; ++v) { rmax[v] = -1e30f; rsum[v] = 0.f; }

  const _Float16* phb = ph_t + (size_t)gb * Nq * C1q;
  const _Float16* gxb = gx + (size_t)gb * C1q * Nq;

  // software pipeline: chunk nc staged in regs while chunk nc-32 computes
  v8h phr[8], gxr[8];
  ld_ph(phb, 0, t, phr);
  ld_gx(gxb, 0, t, gxr);

  for (int nc = 0; nc < Nq; nc += 32) {
    __syncthreads();                           // prior compute done with LDS
    st_ph(lds_ph, t, phr);
    st_gx(lds_gx, t, gxr);
    __syncthreads();
    if (nc + 32 < Nq) {                        // prefetch next chunk (overlaps
      ld_ph(phb, nc + 32, t, phr);             //  with the WMMA work below)
      ld_gx(gxb, nc + 32, t, gxr);
    }

    // S = Q K^T for this wave's 16 rows x 32 cols (already scaled via theta)
    v8f s0 = {}, s1 = {};
#pragma unroll
    for (int s = 0; s < 8; ++s) {
      v16h b0 = load_frag(lds_ph + s * 32, lane, 264);
      v16h b1 = load_frag(lds_ph + 16 * 264 + s * 32, lane, 264);
      s0 = wmma_f16(ath[s], b0, s0);
      s1 = wmma_f16(ath[s], b1, s1);
    }

    // Online softmax: rows live across 16-lane halves in the D layout.
    float p0[8], p1[8], fac[8];
#pragma unroll
    for (int v = 0; v < 8; ++v) {
      float mv = fmaxf(s0[v], s1[v]);
#pragma unroll
      for (int off = 1; off < 16; off <<= 1)
        mv = fmaxf(mv, __shfl_xor(mv, off, 16));
      float nm = fmaxf(rmax[v], mv);
      fac[v] = __expf(rmax[v] - nm);
      p0[v]  = __expf(s0[v] - nm);
      p1[v]  = __expf(s1[v] - nm);
      float cont = p0[v] + p1[v];
#pragma unroll
      for (int off = 1; off < 16; off <<= 1)
        cont += __shfl_xor(cont, off, 16);
      rsum[v] = rsum[v] * fac[v] + cont;
      rmax[v] = nm;
    }
#pragma unroll
    for (int ct = 0; ct < 16; ++ct)
#pragma unroll
      for (int v = 0; v < 8; ++v) yacc[ct][v] *= fac[v];

    // P to LDS (per-wave region), then Y^T += P * gx^T
    _Float16* pws = lds_p + w * 16 * 40;
    {
      int nl = lane & 15, mh = (lane >> 4) * 8;
#pragma unroll
      for (int v = 0; v < 8; ++v) {
        pws[(mh + v) * 40 + nl]      = (_Float16)p0[v];
        pws[(mh + v) * 40 + nl + 16] = (_Float16)p1[v];
      }
    }
    asm volatile("s_wait_dscnt 0" ::: "memory");
    v16h pa = load_frag(pws, lane, 40);
#pragma unroll
    for (int ct = 0; ct < 16; ++ct) {
      v16h bg = load_frag(lds_gx + ct * 16 * 40, lane, 40);
      yacc[ct] = wmma_f16(pa, bg, yacc[ct]);
    }
  }

  float inv[8];
#pragma unroll
  for (int v = 0; v < 8; ++v) inv[v] = 1.0f / rsum[v];
  const int nl = lane & 15, mh = (lane >> 4) * 8;
#pragma unroll
  for (int ct = 0; ct < 16; ++ct) {
    int c = ct * 16 + nl;
#pragma unroll
    for (int v = 0; v < 8; ++v) {
      int m = mw + mh + v;
      y_t[((size_t)gb * Nq + m) * C1q + c] = (_Float16)(yacc[ct][v] * inv[v]);
    }
  }
}

// ------------------------------------------------------ K3: W-conv GEMM
// grid: x = n-tile (36), y = g*B+b (32); block 128. Output channel-major.
__global__ __launch_bounds__(128)
void k3_wconv(const _Float16* __restrict__ y_t, const float* __restrict__ ww,
              const float* __restrict__ wb, float* __restrict__ wy) {
  __shared__ _Float16 lds_w[128 * 40];
  const int gb = blockIdx.y;
  const int g = gb >> 3, b = gb & 7;
  const int n0 = blockIdx.x * 64;
  const int t = threadIdx.x, w = t >> 5, lane = t & 31;
  const int nw = n0 + w * 16;
  const float* Wg = ww + (size_t)g * CGq * C1q;
  const _Float16* yb = y_t + ((size_t)gb * Nq + nw) * C1q;

  v8f acc[8] = {};
  for (int s = 0; s < 8; ++s) {
    v4f wv[8];
#pragma unroll
    for (int e = 0; e < 8; ++e) {              // gather: 8 b128 loads
      int flat = (e * 128 + t) * 4;            // [o 0..127][c 0..31]
      wv[e] = *(const v4f*)(Wg + (size_t)(flat >> 5) * C1q + s * 32 + (flat & 31));
    }
    __syncthreads();
#pragma unroll
    for (int e = 0; e < 8; ++e) {
      int flat = (e * 128 + t) * 4;
      *(v4h*)(lds_w + (flat >> 5) * 40 + (flat & 31)) = cvt4(wv[e]);
    }
    __syncthreads();
    v16h a = load_frag(yb + s * 32, lane, C1q);
#pragma unroll
    for (int j = 0; j < 8; ++j) {
      v16h bb = load_frag(lds_w + j * 16 * 40, lane, 40);
      acc[j] = wmma_f16(a, bb, acc[j]);
    }
  }
  const int nl = lane & 15, mh = (lane >> 4) * 8;
#pragma unroll
  for (int j = 0; j < 8; ++j) {
    int o = j * 16 + nl;
    float bias = wb[g * CGq + o];
#pragma unroll
    for (int v = 0; v < 8; ++v) {
      int n = nw + mh + v;
      wy[((size_t)(g * CGq + o) * Bq + b) * Nq + n] = acc[j][v] + bias;
    }
  }
}

// ------------------------------------------------- BN stats (per channel)
__global__ __launch_bounds__(256)
void k_bn_stats(const float* __restrict__ x, float* __restrict__ stats) {
  const int ch = blockIdx.x;
  const int plane = Bq * Nq;
  const float* p = x + (size_t)ch * plane;
  float s = 0.f, s2 = 0.f;
  for (int i = threadIdx.x; i < plane; i += 256) {
    float v = p[i]; s += v; s2 += v * v;
  }
  __shared__ float ls[256], ls2[256];
  ls[threadIdx.x] = s; ls2[threadIdx.x] = s2;
  __syncthreads();
  for (int o = 128; o > 0; o >>= 1) {
    if (threadIdx.x < o) {
      ls[threadIdx.x]  += ls[threadIdx.x + o];
      ls2[threadIdx.x] += ls2[threadIdx.x + o];
    }
    __syncthreads();
  }
  if (threadIdx.x == 0) {
    float mean = ls[0] * (1.0f / plane);
    float var  = ls2[0] * (1.0f / plane) - mean * mean;
    stats[ch * 2]     = mean;
    stats[ch * 2 + 1] = rsqrtf(var + 1e-5f);
  }
}

// -------------------------------------- K5: BN apply + scatter into ctx
__global__ void k5_bn_to_ctx(const float* __restrict__ wy,
                             const float* __restrict__ stats,
                             const float* __restrict__ gamma,
                             const float* __restrict__ beta,
                             _Float16* __restrict__ ctx) {
  size_t idx = (size_t)blockIdx.x * blockDim.x + threadIdx.x;
  if (idx >= (size_t)Gq * CGq * Bq * Nq) return;
  int n = (int)(idx % Nq);
  size_t r = idx / Nq;
  int b = (int)(r % Bq);
  int ch = (int)(r / Bq);                      // g*CG + o in [0,512)
  float v = (wy[idx] - stats[ch * 2]) * stats[ch * 2 + 1] * gamma[ch] + beta[ch];
  ctx[((size_t)b * Nq + n) * 1024 + Cq + ch] = (_Float16)v;
}

// ---------------------------------------------- K6: bottleneck GEMM
// grid: x = mtile(8) * ntile(36), y = b (8); block 128. B-frags straight
// from global ctx (channel-last => contiguous K per column).
__global__ __launch_bounds__(128)
void k6_bc(const _Float16* __restrict__ ctx, const float* __restrict__ bcw,
           const float* __restrict__ bcb, float* __restrict__ bco) {
  __shared__ _Float16 lds_a[64 * 40];
  const int b = blockIdx.y;
  const int m0 = (blockIdx.x / 36) * 64;
  const int n0 = (blockIdx.x % 36) * 64;
  const int t = threadIdx.x, w = t >> 5, lane = t & 31;
  const int mw = m0 + w * 16;
  const _Float16* cb = ctx + (size_t)b * Nq * 1024;

  v8f acc[4] = {};
  for (int s = 0; s < 32; ++s) {
    v4f av[4];
#pragma unroll
    for (int e = 0; e < 4; ++e) {              // gather: 4 b128 loads
      int flat = (e * 128 + t) * 4;
      av[e] = *(const v4f*)(bcw + (size_t)(m0 + (flat >> 5)) * 1024
                            + s * 32 + (flat & 31));
    }
    __syncthreads();
#pragma unroll
    for (int e = 0; e < 4; ++e) {
      int flat = (e * 128 + t) * 4;
      *(v4h*)(lds_a + (flat >> 5) * 40 + (flat & 31)) = cvt4(av[e]);
    }
    __syncthreads();
    v16h a = load_frag(lds_a + (w * 16) * 40, lane, 40);
#pragma unroll
    for (int j = 0; j < 4; ++j) {
      v16h bb = load_frag(cb + (size_t)(n0 + j * 16) * 1024 + s * 32, lane, 1024);
      acc[j] = wmma_f16(a, bb, acc[j]);
    }
  }
  const int nl = lane & 15, mh = (lane >> 4) * 8;
  float bias_v[8];
#pragma unroll
  for (int v = 0; v < 8; ++v) bias_v[v] = bcb[mw + mh + v];
#pragma unroll
  for (int j = 0; j < 4; ++j) {
    int n = n0 + j * 16 + nl;
#pragma unroll
    for (int v = 0; v < 8; ++v) {
      int O = mw + mh + v;
      bco[((size_t)O * Bq + b) * Nq + n] = acc[j][v] + bias_v[v];
    }
  }
}

// ---------------------------------------------- K8: final BN -> output
__global__ void k8_bn_out(const float* __restrict__ bco,
                          const float* __restrict__ stats,
                          const float* __restrict__ gamma,
                          const float* __restrict__ beta,
                          float* __restrict__ out) {
  size_t idx = (size_t)blockIdx.x * blockDim.x + threadIdx.x;
  if (idx >= (size_t)OUTq * Bq * Nq) return;
  int n = (int)(idx % Nq);
  size_t r = idx / Nq;
  int b = (int)(r % Bq);
  int O = (int)(r / Bq);
  float v = (bco[idx] - stats[O * 2]) * stats[O * 2 + 1] * gamma[O] + beta[O];
  out[((size_t)b * OUTq + O) * Nq + n] = v;
}

extern "C" void kernel_launch(void* const* d_in, const int* in_sizes, int n_in,
                              void* d_out, int out_size, void* d_ws, size_t ws_size,
                              hipStream_t stream) {
  const float* feats    = (const float*)d_in[0];
  const float* theta_w  = (const float*)d_in[1];
  const float* theta_b  = (const float*)d_in[2];
  const float* phi_w    = (const float*)d_in[3];
  const float* phi_b    = (const float*)d_in[4];
  const float* g_w      = (const float*)d_in[5];
  const float* g_b      = (const float*)d_in[6];
  const float* w_w      = (const float*)d_in[7];
  const float* w_b      = (const float*)d_in[8];
  const float* w_gamma  = (const float*)d_in[9];
  const float* w_beta   = (const float*)d_in[10];
  const float* bc_w     = (const float*)d_in[11];
  const float* bc_b     = (const float*)d_in[12];
  const float* bc_gamma = (const float*)d_in[13];
  const float* bc_beta  = (const float*)d_in[14];
  float* out = (float*)d_out;

  const size_t SZ = (size_t)Gq * Bq * Nq * C1q * sizeof(_Float16); // 37,748,736
  char* ws = (char*)d_ws;
  _Float16* th_t = (_Float16*)(ws);
  _Float16* ph_t = (_Float16*)(ws + SZ);
  _Float16* gx   = (_Float16*)(ws + 2 * SZ);
  _Float16* y_t  = (_Float16*)(ws + 3 * SZ);
  _Float16* ctx  = (_Float16*)(ws + 4 * SZ);
  float* wy       = (float*)(ws);              // reuse th_t (dead after K2)
  float* bco      = (float*)(ws + SZ);         // reuse ph_t (dead after K2)
  float* stats_w  = (float*)(ws + 5 * SZ);
  float* stats_bc = (float*)(ws + 5 * SZ + 8192);

  {
    size_t tot = (size_t)Bq * Cq * Nq;
    k0_pack_feats<<<dim3((unsigned)((tot + 255) / 256)), dim3(256), 0, stream>>>(feats, ctx);
  }
  k1_proj<<<dim3(144, 32, 3), dim3(128), 0, stream>>>(
      feats, theta_w, theta_b, phi_w, phi_b, g_w, g_b, th_t, ph_t, gx);
  k2_attn<<<dim3(36, 32), dim3(128), 0, stream>>>(th_t, ph_t, gx, y_t);
  k3_wconv<<<dim3(36, 32), dim3(128), 0, stream>>>(y_t, w_w, w_b, wy);
  k_bn_stats<<<dim3(Gq * CGq), dim3(256), 0, stream>>>(wy, stats_w);
  {
    size_t tot = (size_t)Gq * CGq * Bq * Nq;
    k5_bn_to_ctx<<<dim3((unsigned)((tot + 255) / 256)), dim3(256), 0, stream>>>(
        wy, stats_w, w_gamma, w_beta, ctx);
  }
  k6_bc<<<dim3(288, 8), dim3(128), 0, stream>>>(ctx, bc_w, bc_b, bco);
  k_bn_stats<<<dim3(OUTq), dim3(256), 0, stream>>>(bco, stats_bc);
  {
    size_t tot = (size_t)OUTq * Bq * Nq;
    k8_bn_out<<<dim3((unsigned)((tot + 255) / 256)), dim3(256), 0, stream>>>(
        bco, stats_bc, bc_gamma, bc_beta, out);
  }
  (void)in_sizes; (void)n_in; (void)out_size; (void)ws_size;
}
